// UpdateNodeInFrame_85744727097796
// MI455X (gfx1250) — matrix operand order repaired and checked by reading, not compiled, not run on hardware
//
#include <hip/hip_runtime.h>
#include <math.h>
#include <stdint.h>

// ---------------------------------------------------------------------------
// CDNA5 (gfx1250) wave32 WMMA types
// ---------------------------------------------------------------------------
typedef __attribute__((ext_vector_type(16))) __bf16 v16bf;
typedef __attribute__((ext_vector_type(8)))  __bf16 v8bf;
typedef __attribute__((ext_vector_type(8)))  float  v8f;

#define S_CH 128
#define V_CH 64
#define G_CH 64
#define D_CH 320

__device__ __forceinline__ float sigmoidf_(float x) { return 1.f / (1.f + __expf(-x)); }
__device__ __forceinline__ float siluf_(float x)    { return x * sigmoidf_(x); }

// ---------------------------------------------------------------------------
// WMMA fragment loads (bf16 16x16x32). A is MxK row-major. B is stored
// pre-transposed (N-major: Bt[n*K + k]) so loads are contiguous 16B pairs.
// A 16x32 layout: lanes 0-15 -> M=lane, contiguous K run kh..kh+15,
// kh = (lane>>4)*16 (matches B fragment K order).
// ---------------------------------------------------------------------------
__device__ __forceinline__ v16bf frag_a_load(const __bf16* __restrict__ A, int lda,
                                             int row0, int kb, int M, int lane)
{
    int m = row0 + (lane & 15);
    if (m >= M) m = M - 1;                  // clamp (stores are guarded)
    int kh = (lane >> 4) << 4;
    const __bf16* p = A + (size_t)m * lda + kb + kh;
    v8bf lo = *(const v8bf*)p;
    v8bf hi = *(const v8bf*)(p + 8);
    v16bf r;
#pragma unroll
    for (int i = 0; i < 8; ++i) { r[i] = lo[i]; r[i + 8] = hi[i]; }
    return r;
}

// B 32x16: lane -> column n, contiguous K run of 16 per lane (n-major storage)
__device__ __forceinline__ v16bf frag_b_load(const __bf16* __restrict__ Bt, int ldb,
                                             int col0, int kb, int lane)
{
    int n = col0 + (lane & 15);
    const __bf16* p = Bt + (size_t)n * ldb + kb + ((lane >> 4) << 4);
    v8bf lo = *(const v8bf*)p;
    v8bf hi = *(const v8bf*)(p + 8);
    v16bf r;
#pragma unroll
    for (int i = 0; i < 8; ++i) { r[i] = lo[i]; r[i + 8] = hi[i]; }
    return r;
}

// ---------------------------------------------------------------------------
// Weight convert + transpose: src fp32 (K x N) -> dst bf16 (N x K), * sign
// ---------------------------------------------------------------------------
__global__ void convT_kernel(const float* __restrict__ src, __bf16* __restrict__ dst,
                             int K, int N, float sgn)
{
    int i = blockIdx.x * 256 + threadIdx.x;
    if (i < K * N) {
        int k = i / N, n = i - k * N;
        dst[(size_t)n * K + k] = (__bf16)(src[i] * sgn);
    }
}

// ---------------------------------------------------------------------------
// Node separable layernorm: one wave per node
// ---------------------------------------------------------------------------
__global__ void node_ln_kernel(const float* __restrict__ nodef,
                               const float* __restrict__ nws, const float* __restrict__ nbs,
                               const float* __restrict__ nwv,
                               float* __restrict__ ns, float* __restrict__ nv, int N)
{
    int wslot = threadIdx.x >> 5;
    int lane  = threadIdx.x & 31;
    int n = blockIdx.x * 8 + wslot;
    if (n >= N) return;
    const float* f = nodef + (size_t)n * D_CH;

    float s1 = 0.f, s2 = 0.f;
    for (int i = lane; i < 128; i += 32) { float x = f[i]; s1 += x; s2 += x * x; }
#pragma unroll
    for (int o = 16; o; o >>= 1) { s1 += __shfl_xor(s1, o); s2 += __shfl_xor(s2, o); }
    float mu = s1 * (1.f / 128.f);
    float rstd = rsqrtf(s2 * (1.f / 128.f) - mu * mu + 1e-8f);
    for (int i = lane; i < 128; i += 32)
        ns[(size_t)n * 128 + i] = (f[i] - mu) * rstd * nws[i] + nbs[i];

    float v2 = 0.f;
    for (int i = lane; i < 192; i += 32) { float x = f[128 + i]; v2 += x * x; }
#pragma unroll
    for (int o = 16; o; o >>= 1) v2 += __shfl_xor(v2, o);
    float vr = rsqrtf(v2 * (1.f / 192.f) + 1e-8f);
    for (int i = lane; i < 192; i += 32)
        nv[(size_t)n * 192 + i] = f[128 + i] * vr * nwv[i / 3];
}

// ---------------------------------------------------------------------------
// Edge prep: edge LN, gather, frame, rotate -> bf16 GEMM operands. Wave/edge.
// ---------------------------------------------------------------------------
__global__ void edge_prep_kernel(const float* __restrict__ edgef,
                                 const float* __restrict__ ns, const float* __restrict__ nv,
                                 const int* __restrict__ eidx, const float* __restrict__ evec,
                                 const int* __restrict__ act, const float* __restrict__ lat,
                                 const float* __restrict__ ews, const float* __restrict__ ebs,
                                 const float* __restrict__ ewv,
                                 float* __restrict__ Rout, __bf16* __restrict__ A0,
                                 __bf16* __restrict__ A1, __bf16* __restrict__ A2,
                                 __bf16* __restrict__ latb, int EA, int Etot)
{
    __shared__ float sh_evf[8][192];
    int wslot = threadIdx.x >> 5;
    int lane  = threadIdx.x & 31;
    int e = blockIdx.x * 8 + wslot;
    if (e >= EA) return;
    int ae = act[e];
    int c  = eidx[ae];
    int nb = eidx[Etot + ae];
    const float* ef = edgef + (size_t)e * D_CH;

    // scalar LN stats
    float s1 = 0.f, s2 = 0.f;
    for (int i = lane; i < 128; i += 32) { float x = ef[i]; s1 += x; s2 += x * x; }
#pragma unroll
    for (int o = 16; o; o >>= 1) { s1 += __shfl_xor(s1, o); s2 += __shfl_xor(s2, o); }
    float mu = s1 * (1.f / 128.f);
    float rstd = rsqrtf(s2 * (1.f / 128.f) - mu * mu + 1e-8f);

    // vector LN -> shared evf
    float v2 = 0.f;
    for (int i = lane; i < 192; i += 32) { float x = ef[128 + i]; v2 += x * x; }
#pragma unroll
    for (int o = 16; o; o >>= 1) v2 += __shfl_xor(v2, o);
    float vr = rsqrtf(v2 * (1.f / 192.f) + 1e-8f);
    for (int i = lane; i < 192; i += 32)
        sh_evf[wslot][i] = ef[128 + i] * vr * ewv[i / 3];

    // edge frame (uniform per wave)
    float ax = evec[(size_t)ae * 3 + 0], ay = evec[(size_t)ae * 3 + 1], az = evec[(size_t)ae * 3 + 2];
    float an = rsqrtf(ax * ax + ay * ay + az * az + 1e-8f);
    ax *= an; ay *= an; az *= an;
    float rx, ry, rz;
    if (fabsf(ax) < 0.9f) { rx = 1.f; ry = 0.f; rz = 0.f; } else { rx = 0.f; ry = 1.f; rz = 0.f; }
    float bx = ay * rz - az * ry, by = az * rx - ax * rz, bz = ax * ry - ay * rx;
    float bn = rsqrtf(bx * bx + by * by + bz * bz + 1e-8f);
    bx *= bn; by *= bn; bz *= bn;
    float cx = ay * bz - az * by, cy = az * bx - ax * bz, cz = ax * by - ay * bx;

    if (lane == 0) {
        float* Rp = Rout + (size_t)e * 9;
        Rp[0] = ax; Rp[1] = ay; Rp[2] = az;
        Rp[3] = bx; Rp[4] = by; Rp[5] = bz;
        Rp[6] = cx; Rp[7] = cy; Rp[8] = cz;
    }

    auto getvc = [&](int u, float& x, float& y, float& z) {
        if (u < 64)       { const float* p = nv + (size_t)c * 192 + u * 3;          x = p[0]; y = p[1]; z = p[2]; }
        else if (u < 128) { const float* p = &sh_evf[wslot][(u - 64) * 3];          x = p[0]; y = p[1]; z = p[2]; }
        else              { const float* p = nv + (size_t)nb * 192 + (u - 128) * 3; x = p[0]; y = p[1]; z = p[2]; }
    };

    // m0_in row: [ns[c](128), es(128), ns[nb](128), vf_m0(192)]
    __bf16* a0 = A0 + (size_t)e * 576;
#pragma unroll 2
    for (int j = 0; j < 18; ++j) {
        int i = lane + 32 * j;
        float val;
        if (i < 128)       val = ns[(size_t)c * 128 + i];
        else if (i < 256)  { int k = i - 128; val = (ef[k] - mu) * rstd * ews[k] + ebs[k]; }
        else if (i < 384)  val = ns[(size_t)nb * 128 + (i - 256)];
        else { float x, y, z; getvc(i - 384, x, y, z); val = ax * x + ay * y + az * z; }
        a0[i] = (__bf16)val;
    }
    __bf16* a1 = A1 + (size_t)e * 192;
    __bf16* a2 = A2 + (size_t)e * 192;
#pragma unroll
    for (int j = 0; j < 6; ++j) {
        int i = lane + 32 * j;
        float x, y, z; getvc(i, x, y, z);
        a1[i] = (__bf16)(bx * x + by * y + bz * z);
        a2[i] = (__bf16)(cx * x + cy * y + cz * z);
    }
    const float* lrow = lat + (size_t)ae * 128;
    __bf16* lb = latb + (size_t)e * 128;
#pragma unroll
    for (int j = 0; j < 4; ++j) { int i = lane + 32 * j; lb[i] = (__bf16)lrow[i]; }
}

// ---------------------------------------------------------------------------
// LDS-staged bf16 WMMA GEMM (async global->LDS panel of 32 B-columns).
// Block = 4 waves; each wave computes a 16x32 strip (2 accumulators, A reused).
// C(MxN fp32) = scale * A(MxK) @ Bt(NxK)^T
// ---------------------------------------------------------------------------
template <int K>
__global__ void gemm_lds_kernel(const __bf16* __restrict__ A, const __bf16* __restrict__ Bt,
                                float* __restrict__ C, int M, int N, float scale)
{
    __shared__ __align__(16) __bf16 sB[32 * K];   // 32 columns, n-major
    const int lane = threadIdx.x & 31;
    const int col0 = blockIdx.y * 32;

    // Stage Bt[col0..col0+31][0..K) into LDS via async copy (ASYNCcnt path).
    {
        const char* gp = (const char*)(Bt + (size_t)col0 * K);   // contiguous panel
        uint32_t lbase = (uint32_t)(uintptr_t)sB;                // LDS byte offset
        const int tot = 32 * K * 2;
        for (int off = threadIdx.x * 16; off < tot; off += 128 * 16) {
            uint32_t lds_off = lbase + (uint32_t)off;
            uint64_t ga = (uint64_t)(uintptr_t)(gp + off);
            asm volatile("global_load_async_to_lds_b128 %0, %1, off"
                         :: "v"(lds_off), "v"(ga) : "memory");
        }
        asm volatile("s_wait_asynccnt 0" ::: "memory");
        __syncthreads();
    }

    int mt = blockIdx.x * 4 + (threadIdx.x >> 5);
    int row0 = mt * 16;
    if (row0 >= M) row0 = (M - 1) & ~15;    // clamp: duplicate plain stores are idempotent

    v8f acc0 = {}, acc1 = {};
    for (int kb = 0; kb < K; kb += 32) {
        v16bf a = frag_a_load(A, K, row0, kb, M, lane);
        const int nrel = lane & 15;
        const int kh = (lane >> 4) << 4;
        const __bf16* p0 = sB + nrel * K + kb + kh;
        const __bf16* p1 = sB + (nrel + 16) * K + kb + kh;
        v8bf l0 = *(const v8bf*)p0, h0 = *(const v8bf*)(p0 + 8);
        v8bf l1 = *(const v8bf*)p1, h1 = *(const v8bf*)(p1 + 8);
        v16bf b0, b1;
#pragma unroll
        for (int i = 0; i < 8; ++i) {
            b0[i] = l0[i]; b0[i + 8] = h0[i];
            b1[i] = l1[i]; b1[i + 8] = h1[i];
        }
        acc0 = __builtin_amdgcn_wmma_f32_16x16x32_bf16(false, a, false, b0, (short)0, acc0, false, false);
        acc1 = __builtin_amdgcn_wmma_f32_16x16x32_bf16(false, a, false, b1, (short)0, acc1, false, false);
    }
    int col = col0 + (lane & 15);
    int rb  = row0 + ((lane >> 4) << 3);
#pragma unroll
    for (int v = 0; v < 8; ++v) {
        int r = rb + v;
        if (r < M) {
            C[(size_t)r * N + col]      = acc0[v] * scale;
            C[(size_t)r * N + col + 16] = acc1[v] * scale;
        }
    }
}

// ---------------------------------------------------------------------------
// Fused SO(2) m=1 GEMMs: vp = (A1@Wr + A2@(-Wi)) * s ; vm = (A1@Wi + A2@Wr) * s
// ---------------------------------------------------------------------------
__global__ void gemm_vpvm_kernel(const __bf16* __restrict__ A1, const __bf16* __restrict__ A2,
                                 const __bf16* __restrict__ Brt, const __bf16* __restrict__ Bit,
                                 const __bf16* __restrict__ Bint,
                                 float* __restrict__ VP, float* __restrict__ VM,
                                 int M, float scale)
{
    const int K = 192, N = 64;
    int mt = blockIdx.x * 4 + (threadIdx.x >> 5);
    int row0 = mt * 16;
    if (row0 >= M) return;
    int nt = blockIdx.y;
    const int lane = threadIdx.x & 31;
    v8f accP = {}, accM = {};
    for (int kb = 0; kb < K; kb += 32) {
        v16bf a1 = frag_a_load(A1, K, row0, kb, M, lane);
        v16bf a2 = frag_a_load(A2, K, row0, kb, M, lane);
        v16bf br = frag_b_load(Brt,  K, nt * 16, kb, lane);
        v16bf bi = frag_b_load(Bit,  K, nt * 16, kb, lane);
        v16bf bn = frag_b_load(Bint, K, nt * 16, kb, lane);
        accP = __builtin_amdgcn_wmma_f32_16x16x32_bf16(false, a1, false, br, (short)0, accP, false, false);
        accP = __builtin_amdgcn_wmma_f32_16x16x32_bf16(false, a2, false, bn, (short)0, accP, false, false);
        accM = __builtin_amdgcn_wmma_f32_16x16x32_bf16(false, a1, false, bi, (short)0, accM, false, false);
        accM = __builtin_amdgcn_wmma_f32_16x16x32_bf16(false, a2, false, br, (short)0, accM, false, false);
    }
    int col = nt * 16 + (lane & 15);
    int rb  = row0 + ((lane >> 4) << 3);
#pragma unroll
    for (int v = 0; v < 8; ++v) {
        int r = rb + v;
        if (r < M) {
            VP[(size_t)r * N + col] = accP[v] * scale;
            VM[(size_t)r * N + col] = accM[v] * scale;
        }
    }
}

// ---------------------------------------------------------------------------
// Edge post: rotate back, gate -> bf16 operands for the lp linears. Wave/edge.
// ---------------------------------------------------------------------------
__global__ void edge_post_kernel(const float* __restrict__ m0o, const float* __restrict__ VP,
                                 const float* __restrict__ VM, const float* __restrict__ Rb,
                                 __bf16* __restrict__ msb, __bf16* __restrict__ mvb, int EA)
{
    int wslot = threadIdx.x >> 5;
    int lane  = threadIdx.x & 31;
    int e = blockIdx.x * 8 + wslot;
    if (e >= EA) return;
    const float* m0 = m0o + (size_t)e * 256;
    const float* Rp = Rb + (size_t)e * 9;
    float r0 = Rp[0], r1 = Rp[1], r2 = Rp[2];
    float r3 = Rp[3], r4 = Rp[4], r5 = Rp[5];
    float r6 = Rp[6], r7 = Rp[7], r8 = Rp[8];

#pragma unroll
    for (int j = 0; j < 2; ++j) {
        int u = lane + 32 * j;   // 0..63 vector channels
        float v0 = m0[192 + u];
        float v1 = VP[(size_t)e * 64 + u];
        float v2 = VM[(size_t)e * 64 + u];
        // back-rotate: out[j] = sum_i R[i][j] * v[i]
        float ox = r0 * v0 + r3 * v1 + r6 * v2;
        float oy = r1 * v0 + r4 * v1 + r7 * v2;
        float oz = r2 * v0 + r5 * v1 + r8 * v2;
        float g = sigmoidf_(m0[128 + u]);
        mvb[((size_t)0 * EA + e) * 64 + u] = (__bf16)(g * ox);
        mvb[((size_t)1 * EA + e) * 64 + u] = (__bf16)(g * oy);
        mvb[((size_t)2 * EA + e) * 64 + u] = (__bf16)(g * oz);
    }
#pragma unroll
    for (int j = 0; j < 4; ++j) {
        int i = lane + 32 * j;   // 0..127 scalar channels
        msb[(size_t)e * 128 + i] = (__bf16)siluf_(m0[i]);
    }
}

// ---------------------------------------------------------------------------
// lp scalar linear + env weight + segment-sum aggregation (atomic epilogue)
// ---------------------------------------------------------------------------
__global__ void gemm_agg_s_kernel(const __bf16* __restrict__ A, const __bf16* __restrict__ Bt,
                                  const float* __restrict__ bias, const float* __restrict__ wenv,
                                  const int* __restrict__ eidx, const int* __restrict__ act,
                                  float* __restrict__ aggs, int M, int Etot)
{
    const int K = 128, N = 128;
    const float rs128 = 0.08838834764831845f;   // 1/sqrt(128)
    const float inv_avg = 0.2886751345948129f;  // 1/sqrt(12)
    int mt = blockIdx.x * 4 + (threadIdx.x >> 5);
    int row0 = mt * 16;
    if (row0 >= M) return;
    int nt = blockIdx.y;
    const int lane = threadIdx.x & 31;
    v8f acc = {};
    for (int kb = 0; kb < K; kb += 32) {
        v16bf a = frag_a_load(A, K, row0, kb, M, lane);
        v16bf b = frag_b_load(Bt, K, nt * 16, kb, lane);
        acc = __builtin_amdgcn_wmma_f32_16x16x32_bf16(false, a, false, b, (short)0, acc, false, false);
    }
    int col = nt * 16 + (lane & 15);
    int rb  = row0 + ((lane >> 4) << 3);
    float bcol = bias[col];
#pragma unroll
    for (int v = 0; v < 8; ++v) {
        int r = rb + v;
        if (r < M) {
            float val = acc[v] * rs128 + bcol;
            val *= wenv[(size_t)r * 192 + col];
            int ct = eidx[act[r]];
            atomicAdd(&aggs[(size_t)ct * 128 + col], val * inv_avg);
        }
    }
}

// lp vector linear (3 planes via grid.z) + env weight + aggregation
__global__ void gemm_agg_v_kernel(const __bf16* __restrict__ Ap, const __bf16* __restrict__ Bt,
                                  const float* __restrict__ wenv,
                                  const int* __restrict__ eidx, const int* __restrict__ act,
                                  float* __restrict__ aggv, int M, int Etot)
{
    const int K = 64, N = 64;
    const float rs64 = 0.125f;                  // 1/sqrt(64)
    const float inv_avg = 0.2886751345948129f;  // 1/sqrt(12)
    int m = blockIdx.z;
    const __bf16* A = Ap + (size_t)m * M * 64;
    int mt = blockIdx.x * 4 + (threadIdx.x >> 5);
    int row0 = mt * 16;
    if (row0 >= M) return;
    int nt = blockIdx.y;
    const int lane = threadIdx.x & 31;
    v8f acc = {};
    for (int kb = 0; kb < K; kb += 32) {
        v16bf a = frag_a_load(A, K, row0, kb, M, lane);
        v16bf b = frag_b_load(Bt, K, nt * 16, kb, lane);
        acc = __builtin_amdgcn_wmma_f32_16x16x32_bf16(false, a, false, b, (short)0, acc, false, false);
    }
    int col = nt * 16 + (lane & 15);
    int rb  = row0 + ((lane >> 4) << 3);
#pragma unroll
    for (int v = 0; v < 8; ++v) {
        int r = rb + v;
        if (r < M) {
            float val = acc[v] * rs64 * wenv[(size_t)r * 192 + 128 + col];
            int ct = eidx[act[r]];
            atomicAdd(&aggv[((size_t)ct * 64 + col) * 3 + m], val * inv_avg);
        }
    }
}

// ---------------------------------------------------------------------------
// Per-node: atom-type-sliced contractions, gate, output linear, residual
// ---------------------------------------------------------------------------
__global__ void node_out_kernel(const float* __restrict__ nodef, const int* __restrict__ atype,
                                const float* __restrict__ aggs, const float* __restrict__ aggv,
                                const float* __restrict__ ohss, const float* __restrict__ ohsg,
                                const float* __restrict__ ohvv,
                                const float* __restrict__ ohlws, const float* __restrict__ ohlbs,
                                const float* __restrict__ ohlwv,
                                const float* __restrict__ resp, float* __restrict__ out, int N)
{
    __shared__ float sA[128];
    __shared__ float sV[192];
    __shared__ float sGs[128];
    __shared__ float sGv[192];
    int n = blockIdx.x;
    int k = threadIdx.x;
    int t = atype[n];
    if (t < 0) t = 0;
    if (t > 63) t = 63;
    sA[k] = aggs[(size_t)n * 128 + k];
    for (int i = k; i < 192; i += 128) sV[i] = aggv[(size_t)n * 192 + i];
    __syncthreads();

    const float n_ss = 0.011048543456039805f;  // 1/sqrt(128*64)
    const float n_vv = 0.015625f;              // 1/sqrt(64*64)
    {
        const float* W = ohss + (size_t)t * 128 + k;
        float acc = 0.f;
        for (int u = 0; u < 128; ++u) acc += sA[u] * W[(size_t)u * 8192];
        sGs[k] = siluf_(acc * n_ss);
    }
    if (k < 64) {
        const float* Wg = ohsg + (size_t)t * 64 + k;
        float tg = 0.f;
        for (int u = 0; u < 128; ++u) tg += sA[u] * Wg[(size_t)u * 4096];
        const float* Wv = ohvv + (size_t)t * 64 + k;
        float t0 = 0.f, t1 = 0.f, t2 = 0.f;
        for (int u = 0; u < 64; ++u) {
            float w = Wv[(size_t)u * 4096];
            t0 += sV[u * 3 + 0] * w; t1 += sV[u * 3 + 1] * w; t2 += sV[u * 3 + 2] * w;
        }
        float g = sigmoidf_(tg * n_ss);
        sGv[k * 3 + 0] = g * t0 * n_vv;
        sGv[k * 3 + 1] = g * t1 * n_vv;
        sGv[k * 3 + 2] = g * t2 * n_vv;
    }
    __syncthreads();

    float alpha = sigmoidf_(resp[0]);
    float beta  = 1.f - alpha;
    const float rs128 = 0.08838834764831845f;
    float* orow = out + (size_t)n * D_CH;
    const float* frow = nodef + (size_t)n * D_CH;
    {
        float acc = 0.f;
        for (int u = 0; u < 128; ++u) acc += sGs[u] * ohlws[u * 128 + k];
        float os = acc * rs128 + ohlbs[k];
        orow[k] = alpha * frow[k] + beta * os;
    }
    if (k < 64) {
        float o0 = 0.f, o1 = 0.f, o2 = 0.f;
        for (int u = 0; u < 64; ++u) {
            float w = ohlwv[u * 64 + k];
            o0 += sGv[u * 3 + 0] * w; o1 += sGv[u * 3 + 1] * w; o2 += sGv[u * 3 + 2] * w;
        }
        int b = 128 + k * 3;
        orow[b + 0] = alpha * frow[b + 0] + beta * (o0 * 0.125f);
        orow[b + 1] = alpha * frow[b + 1] + beta * (o1 * 0.125f);
        orow[b + 2] = alpha * frow[b + 2] + beta * (o2 * 0.125f);
    }
}

// ---------------------------------------------------------------------------
// Launcher
// ---------------------------------------------------------------------------
extern "C" void kernel_launch(void* const* d_in, const int* in_sizes, int n_in,
                              void* d_out, int out_size, void* d_ws, size_t ws_size,
                              hipStream_t stream)
{
    const float* latents = (const float*)d_in[0];
    const float* nodef   = (const float*)d_in[1];
    const float* edgef   = (const float*)d_in[2];
    const int*   atype   = (const int*)d_in[3];
    const int*   eidx    = (const int*)d_in[5];
    const float* evec    = (const float*)d_in[6];
    const int*   act     = (const int*)d_in[7];
    const float* nws = (const float*)d_in[8];
    const float* nbs = (const float*)d_in[9];
    const float* nwv = (const float*)d_in[10];
    const float* ews = (const float*)d_in[11];
    const float* ebs = (const float*)d_in[12];
    const float* ewv = (const float*)d_in[13];
    const float* w0   = (const float*)d_in[14];
    const float* w1r  = (const float*)d_in[15];
    const float* w1i  = (const float*)d_in[16];
    const float* envw = (const float*)d_in[17];
    const float* lpws = (const float*)d_in[18];
    const float* lpbs = (const float*)d_in[19];
    const float* lpwv = (const float*)d_in[20];
    const float* ohss = (const float*)d_in[21];
    const float* ohsg = (const float*)d_in[22];
    const float* ohvv = (const float*)d_in[23];
    const float* ohlws = (const float*)d_in[24];
    const float* ohlbs = (const float*)d_in[25];
    const float* ohlwv = (const float*)d_in[26];
    const float* resp  = (const float*)d_in[27];

    const int N    = in_sizes[1] / D_CH;
    const int Etot = in_sizes[2] / D_CH;
    const int EA   = in_sizes[7];

    char* wp = (char*)d_ws;
    auto carve = [&](size_t bytes) -> char* {
        char* p = wp;
        wp += (bytes + 255) & ~(size_t)255;
        return p;
    };
    float*  ns   = (float*)carve((size_t)N * 128 * 4);
    float*  nv   = (float*)carve((size_t)N * 192 * 4);
    float*  Rb   = (float*)carve((size_t)EA * 9 * 4);
    __bf16* A0   = (__bf16*)carve((size_t)EA * 576 * 2);
    __bf16* A1   = (__bf16*)carve((size_t)EA * 192 * 2);
    __bf16* A2   = (__bf16*)carve((size_t)EA * 192 * 2);
    __bf16* latb = (__bf16*)carve((size_t)EA * 128 * 2);
    __bf16* W0t  = (__bf16*)carve((size_t)576 * 256 * 2);
    __bf16* Brt  = (__bf16*)carve((size_t)192 * 64 * 2);
    __bf16* Bit  = (__bf16*)carve((size_t)192 * 64 * 2);
    __bf16* Bint = (__bf16*)carve((size_t)192 * 64 * 2);
    __bf16* Et   = (__bf16*)carve((size_t)128 * 192 * 2);
    __bf16* Lst  = (__bf16*)carve((size_t)128 * 128 * 2);
    __bf16* Lvt  = (__bf16*)carve((size_t)64 * 64 * 2);
    float*  m0o  = (float*)carve((size_t)EA * 256 * 4);
    float*  vpb  = (float*)carve((size_t)EA * 64 * 4);
    float*  vmb  = (float*)carve((size_t)EA * 64 * 4);
    float*  aggs = (float*)carve((size_t)N * 128 * 4);
    float*  aggv = (float*)carve((size_t)N * 192 * 4);
    // Lifetime reuse (same-stream ordering guarantees safety):
    float*  wenv = (float*)A1;                 // EA*192 f32 over [A1..A2] (dead after vp/vm GEMM)
    __bf16* msb  = A0;                         // EA*128 bf16 in A0 region (dead after m0 GEMM)
    __bf16* mvb  = A0 + (size_t)EA * 128;      // 3 planes of EA*64 bf16

    hipMemsetAsync(aggs, 0, (size_t)N * 128 * 4, stream);
    hipMemsetAsync(aggv, 0, (size_t)N * 192 * 4, stream);

    auto cvt = [&](const float* src, __bf16* dst, int K, int Ncols, float sgn) {
        int tot = K * Ncols;
        convT_kernel<<<(tot + 255) / 256, 256, 0, stream>>>(src, dst, K, Ncols, sgn);
    };
    cvt(w0,   W0t,  576, 256,  1.f);
    cvt(w1r,  Brt,  192, 64,   1.f);
    cvt(w1i,  Bit,  192, 64,   1.f);
    cvt(w1i,  Bint, 192, 64,  -1.f);
    cvt(envw, Et,   128, 192,  1.f);
    cvt(lpws, Lst,  128, 128,  1.f);
    cvt(lpwv, Lvt,  64,  64,   1.f);

    node_ln_kernel<<<(N + 7) / 8, 256, 0, stream>>>(nodef, nws, nbs, nwv, ns, nv, N);
    edge_prep_kernel<<<(EA + 7) / 8, 256, 0, stream>>>(edgef, ns, nv, eidx, evec, act, latents,
                                                       ews, ebs, ewv, Rb, A0, A1, A2, latb,
                                                       EA, Etot);

    int mtiles = (EA + 15) / 16;
    int mblk = (mtiles + 3) / 4;
    dim3 blk(128);
    // m0 GEMM: K=576, N=256 -> 8 panels of 32 columns, LDS-staged via async copy
    gemm_lds_kernel<576><<<dim3(mblk, 8), blk, 0, stream>>>(A0, W0t, m0o, EA, 256,
                                                            1.f / sqrtf(576.f));
    gemm_vpvm_kernel<<<dim3(mblk, 4), blk, 0, stream>>>(A1, A2, Brt, Bit, Bint, vpb, vmb, EA,
                                                        1.f / sqrtf(192.f));
    // env GEMM: K=128, N=192 -> 6 panels of 32 columns
    gemm_lds_kernel<128><<<dim3(mblk, 6), blk, 0, stream>>>(latb, Et, wenv, EA, 192,
                                                            1.f / sqrtf(128.f));

    edge_post_kernel<<<(EA + 7) / 8, 256, 0, stream>>>(m0o, vpb, vmb, Rb, msb, mvb, EA);

    gemm_agg_s_kernel<<<dim3(mblk, 8), blk, 0, stream>>>(msb, Lst, lpbs, wenv, eidx, act,
                                                         aggs, EA, Etot);
    gemm_agg_v_kernel<<<dim3(mblk, 4, 3), blk, 0, stream>>>(mvb, Lvt, wenv, eidx, act,
                                                            aggv, EA, Etot);

    node_out_kernel<<<N, 128, 0, stream>>>(nodef, atype, aggs, aggv, ohss, ohsg, ohvv,
                                           ohlws, ohlbs, ohlwv, resp, (float*)d_out, N);
}
